// GnnRnnAgent_39719857554102
// MI455X (gfx1250) — matrix operand order repaired and checked by reading, not compiled, not run on hardware
//
#include <hip/hip_runtime.h>
#include <hip/hip_bf16.h>
#include <math.h>

#define NN    65536
#define EE    524288
#define DIN   128
#define HH    128
#define OUTD  256
#define NA    16
#define EF    (EE + NN)
#define SLOPE 0.2f

typedef __attribute__((ext_vector_type(16))) __bf16 v16bf;
typedef __attribute__((ext_vector_type(8)))  float  v8f;

// ---------------------------------------------------------------- WMMA core
__device__ __forceinline__ v8f wmma_bf16(v16bf a, v16bf b, v8f c) {
  return __builtin_amdgcn_wmma_f32_16x16x32_bf16(false, a, false, b, (short)0, c, false, false);
}

// ---- fragment-major packed layouts (ISA 7.12.2 wave32) ----
// A 16x32 tile, lane L: row=L%16, kb=8*(L>=16); elem e: kk=(e&7)+8*(L>=16)+16*(e>>3)
// packed A tile t: offset = t*512 + lane*16 + e  (16 contiguous bf16 per lane = 32B)
// B 32x16 tile, lane L: n=L%16, kb=8*(L>=16); elem e: kk=(e&7)+kb+16*(e>>3)
// packed B (t,j): offset = ((t*numJ + j)*32 + lane)*16 + e

__device__ __forceinline__ v16bf load_Ap(const __bf16* P, int t, int lane) {
  return *(const v16bf*)(P + t * 512 + lane * 16);
}
__device__ __forceinline__ v16bf load_Bp(const __bf16* P, int numJ, int t, int j, int lane) {
  return *(const v16bf*)(P + ((size_t)(t * numJ + j) * 32 + lane) * 16);
}
// (row, kk in [0,32)) -> packed offset within one A k-tile
__device__ __forceinline__ int a_off(int row, int kk) {
  int hi = (kk >> 3) & 1;
  int e  = (kk & 7) + ((kk >> 4) << 3);
  return (row + (hi << 4)) * 16 + e;
}

__device__ __forceinline__ void atomicMaxF(float* addr, float v) {
  if (v >= 0.f) atomicMax((int*)addr, __float_as_int(v));
  else          atomicMin((unsigned int*)addr, __float_as_uint(v));
}
__device__ __forceinline__ float sigmoidf(float x) { return 1.f / (1.f + __expf(-x)); }

// ------------------------------------------------- weight pack (f32 -> bf16, fragment-major)
// src: K x N row-major (k-major). transpose=1: src is N x K row-major (x @ W^T case).
__global__ void k_pack_B(const float* __restrict__ src, __bf16* __restrict__ dst,
                         int K, int N, int transpose) {
  int i = blockIdx.x * blockDim.x + threadIdx.x;
  if (i >= K * N) return;
  int numJ = N >> 4;
  int e    = i & 15;
  int lane = (i >> 4) & 31;
  int j    = (i >> 9) % numJ;
  int t    = (i >> 9) / numJ;
  int n    = lane & 15;
  int kb   = (lane >> 4) << 3;
  int k    = t * 32 + kb + (e & 7) + ((e >> 3) << 4);
  int col  = j * 16 + n;
  float v = transpose ? src[(size_t)col * K + k] : src[(size_t)k * N + col];
  dst[i] = (__bf16)v;
}

__global__ void k_init(float* agg, float* deg, float* loop3, float* smax, float* den) {
  int i = blockIdx.x * blockDim.x + threadIdx.x;   // grid covers N*OUTD
  agg[i] = 0.f;
  if (i < NN)     { deg[i] = 0.f; den[i] = 0.f; smax[i] = __int_as_float(0xFF800000); }
  if (i < 3 * NN) loop3[i] = 0.f;
}

// ---------------------------------------------------------------- node MLP + GAT linears
__global__ void __launch_bounds__(32)
k_node(const float* __restrict__ inputs,
       const __bf16* __restrict__ W0p, const float* __restrict__ b0,
       const float* __restrict__ g0, const float* __restrict__ be0,
       const __bf16* __restrict__ W1p, const float* __restrict__ b1,
       const float* __restrict__ g1, const float* __restrict__ be1,
       const __bf16* __restrict__ Wlp, const float* __restrict__ bl,
       const __bf16* __restrict__ Wrp, const float* __restrict__ br,
       const __bf16* __restrict__ Wresp,
       float* __restrict__ x_l, float* __restrict__ x_r, float* __restrict__ xres) {
  __shared__ __attribute__((aligned(32))) __bf16 sAp[16 * HH];   // packed A, 4 k-tiles
  __shared__ float sX[16 * HH];                                   // row-major f32
  const int lane = threadIdx.x;
  const size_t row0 = (size_t)blockIdx.x * 16;

  // stage input tile straight into packed-A layout
  for (int i = lane; i < 16 * DIN; i += 32) {
    int e = i & 15, ln = (i >> 4) & 31, t = i >> 9;
    int row = ln & 15;
    int kk  = (e & 7) + ((ln >> 4) << 3) + ((e >> 3) << 4);
    sAp[i] = (__bf16)inputs[(row0 + row) * DIN + t * 32 + kk];
  }
  __syncthreads();

  // ---- two MLP layers: GEMM(16x128 @ 128x128) + bias + relu + LN ----
  for (int layer = 0; layer < 2; ++layer) {
    const __bf16* Wp = layer ? W1p : W0p;
    const float *bi = layer ? b1 : b0, *ga = layer ? g1 : g0, *bb = layer ? be1 : be0;
    v16bf af[4];
    #pragma unroll
    for (int t = 0; t < 4; ++t) af[t] = load_Ap(sAp, t, lane);
    for (int j = 0; j < 8; ++j) {
      v8f acc = {};
      #pragma unroll
      for (int t = 0; t < 4; ++t)
        acc = wmma_bf16(af[t], load_Bp(Wp, 8, t, j, lane), acc);
      const int col = j * 16 + (lane & 15);
      const float bias = bi[col];
      #pragma unroll
      for (int r = 0; r < 8; ++r) {
        int m = r + ((lane >> 4) << 3);
        float v = acc[r] + bias;
        sX[m * HH + col] = v > 0.f ? v : 0.f;
      }
    }
    __syncthreads();
    // LayerNorm per row; write normalized bf16 back in packed-A layout
    {
      const int row = lane & 15, base = (lane >> 4) * 64;
      float s = 0.f, s2 = 0.f;
      for (int c = 0; c < 64; ++c) { float v = sX[row * HH + base + c]; s += v; s2 += v * v; }
      s  += __shfl_xor(s, 16, 32);
      s2 += __shfl_xor(s2, 16, 32);
      float mean = s * (1.f / HH);
      float var  = s2 * (1.f / HH) - mean * mean;
      float rstd = rsqrtf(var + 1e-5f);
      for (int c = 0; c < 64; ++c) {
        int col = base + c;
        float v = (sX[row * HH + col] - mean) * rstd * ga[col] + bb[col];
        sAp[(col >> 5) * 512 + a_off(row, col & 31)] = (__bf16)v;
      }
    }
    __syncthreads();
  }

  // ---- x_l / x_r / x@Wres : GEMM(16x128 @ 128x256) ----
  v16bf af[4];
  #pragma unroll
  for (int t = 0; t < 4; ++t) af[t] = load_Ap(sAp, t, lane);
  for (int j = 0; j < 16; ++j) {
    v8f aL = {}, aR = {}, aS = {};
    #pragma unroll
    for (int t = 0; t < 4; ++t) {
      aL = wmma_bf16(af[t], load_Bp(Wlp,   16, t, j, lane), aL);
      aR = wmma_bf16(af[t], load_Bp(Wrp,   16, t, j, lane), aR);
      aS = wmma_bf16(af[t], load_Bp(Wresp, 16, t, j, lane), aS);
    }
    const int col = j * 16 + (lane & 15);
    #pragma unroll
    for (int r = 0; r < 8; ++r) {
      int m = r + ((lane >> 4) << 3);
      size_t o = (row0 + m) * OUTD + col;
      x_l[o]  = aL[r] + bl[col];
      x_r[o]  = aR[r] + br[col];
      xres[o] = aS[r];
    }
  }
}

// ---------------------------------------------------------------- edge phase
__global__ void k_edge_prep(const int* __restrict__ ei, const float* __restrict__ eattr,
                            float* __restrict__ deg, float* __restrict__ loop3) {
  int e = blockIdx.x * blockDim.x + threadIdx.x;
  if (e >= EE) return;
  int d = ei[EE + e];
  atomicAdd(&deg[d], 1.0f);
  atomicAdd(&loop3[d * 3 + 0], eattr[e * 3 + 0]);
  atomicAdd(&loop3[d * 3 + 1], eattr[e * 3 + 1]);
  atomicAdd(&loop3[d * 3 + 2], eattr[e * 3 + 2]);
}

__global__ void k_loop_fin(float* __restrict__ loop3, const float* __restrict__ deg) {
  int i = blockIdx.x * blockDim.x + threadIdx.x;
  if (i >= NN) return;
  float inv = 1.f / fmaxf(deg[i], 1.f);
  loop3[i * 3 + 0] *= inv; loop3[i * 3 + 1] *= inv; loop3[i * 3 + 2] *= inv;
}

// one wave32 per edge: score = leaky_relu(x_l[s]+x_r[d]+attr@We) . att ; segment max
__global__ void k_score(const int* __restrict__ ei, const float* __restrict__ eattr,
                        const float* __restrict__ loop3,
                        const float* __restrict__ x_l, const float* __restrict__ x_r,
                        const float* __restrict__ We, const float* __restrict__ att,
                        float* __restrict__ scoreb, float* __restrict__ smax) {
  int e = blockIdx.x * (blockDim.x >> 5) + (threadIdx.x >> 5);
  int lane = threadIdx.x & 31;
  if (e >= EF) return;
  int s, d; float a0, a1, a2;
  if (e < EE) { s = ei[e]; d = ei[EE + e]; a0 = eattr[e*3]; a1 = eattr[e*3+1]; a2 = eattr[e*3+2]; }
  else        { s = d = e - EE; a0 = loop3[s*3]; a1 = loop3[s*3+1]; a2 = loop3[s*3+2]; }
  const float* xl = x_l + (size_t)s * OUTD;
  const float* xr = x_r + (size_t)d * OUTD;
  float p = 0.f;
  #pragma unroll
  for (int t = 0; t < 8; ++t) {
    int j = lane + 32 * t;
    float m = xl[j] + xr[j] + a0 * We[j] + a1 * We[OUTD + j] + a2 * We[2 * OUTD + j];
    float lm = m > 0.f ? m : SLOPE * m;
    p += lm * att[j];
  }
  #pragma unroll
  for (int off = 16; off > 0; off >>= 1) p += __shfl_xor(p, off, 32);
  if (lane == 0) { scoreb[e] = p; atomicMaxF(&smax[d], p); }
}

__global__ void k_exp(const int* __restrict__ ei, float* __restrict__ scoreb,
                      const float* __restrict__ smax, float* __restrict__ den) {
  int e = blockIdx.x * blockDim.x + threadIdx.x;
  if (e >= EF) return;
  int d = (e < EE) ? ei[EE + e] : (e - EE);
  float ex = __expf(scoreb[e] - smax[d]);
  scoreb[e] = ex;
  atomicAdd(&den[d], ex);
}

// one wave32 per edge: agg[d] += alpha * x_l[s]
__global__ void k_agg(const int* __restrict__ ei, const float* __restrict__ scoreb,
                      const float* __restrict__ den, const float* __restrict__ x_l,
                      float* __restrict__ agg) {
  int e = blockIdx.x * (blockDim.x >> 5) + (threadIdx.x >> 5);
  int lane = threadIdx.x & 31;
  if (e >= EF) return;
  int s, d;
  if (e < EE) { s = ei[e]; d = ei[EE + e]; } else { s = d = e - EE; }
  float alpha = scoreb[e] / (den[d] + 1e-16f);
  const float* xl = x_l + (size_t)s * OUTD;
  float* ag = agg + (size_t)d * OUTD;
  #pragma unroll
  for (int t = 0; t < 8; ++t) {
    int j = lane + 32 * t;
    atomicAdd(&ag[j], alpha * xl[j]);
  }
}

__global__ void k_gat(const float* __restrict__ agg, const float* __restrict__ xres,
                      const float* __restrict__ b_gat, __bf16* __restrict__ gatb) {
  int j = threadIdx.x;
  size_t i = (size_t)blockIdx.x * OUTD + j;
  float v = agg[i] + xres[i] + b_gat[j];
  gatb[i] = (__bf16)(v > 0.f ? v : 0.f);
}

// ---------------------------------------------------------------- GRU + head
__global__ void __launch_bounds__(32)
k_gru(const __bf16* __restrict__ gatb, const float* __restrict__ hidden,
      const __bf16* __restrict__ Wihp, const float* __restrict__ b_ih,
      const __bf16* __restrict__ Whhp, const float* __restrict__ b_hh,
      const float* __restrict__ g_out, const float* __restrict__ be_out,
      const __bf16* __restrict__ Wqp, const float* __restrict__ bq,
      float* __restrict__ q_out, float* __restrict__ h_out) {
  __shared__ __attribute__((aligned(32))) __bf16 sGp[16 * OUTD];  // packed A (8 k-tiles)
  __shared__ __attribute__((aligned(32))) __bf16 sHp[16 * OUTD];
  __shared__ float sHo[16 * OUTD];
  const int lane = threadIdx.x;
  const size_t row0 = (size_t)blockIdx.x * 16;

  // stage gat + hidden straight into packed-A layout
  for (int i = lane; i < 16 * OUTD; i += 32) {
    int e = i & 15, ln = (i >> 4) & 31, t = i >> 9;
    int row = ln & 15;
    int kk  = (e & 7) + ((ln >> 4) << 3) + ((e >> 3) << 4);
    size_t src = (row0 + row) * OUTD + t * 32 + kk;
    sGp[i] = gatb[src];
    sHp[i] = (__bf16)hidden[src];
  }
  __syncthreads();

  // hold all A fragments in VGPRs across the column loop
  v16bf agf[8], ahf[8];
  #pragma unroll
  for (int t = 0; t < 8; ++t) { agf[t] = load_Ap(sGp, t, lane); ahf[t] = load_Ap(sHp, t, lane); }

  // per 16-col group: 6 WMMA chains (K=256) then fused gate math  (numJ = 768/16 = 48)
  for (int c = 0; c < 16; ++c) {
    v8f ir = {}, iz = {}, in = {}, hr = {}, hz = {}, hn = {};
    #pragma unroll
    for (int t = 0; t < 8; ++t) {
      ir = wmma_bf16(agf[t], load_Bp(Wihp, 48, t, c,      lane), ir);
      iz = wmma_bf16(agf[t], load_Bp(Wihp, 48, t, 16 + c, lane), iz);
      in = wmma_bf16(agf[t], load_Bp(Wihp, 48, t, 32 + c, lane), in);
      hr = wmma_bf16(ahf[t], load_Bp(Whhp, 48, t, c,      lane), hr);
      hz = wmma_bf16(ahf[t], load_Bp(Whhp, 48, t, 16 + c, lane), hz);
      hn = wmma_bf16(ahf[t], load_Bp(Whhp, 48, t, 32 + c, lane), hn);
    }
    const int col = c * 16 + (lane & 15);
    const float bir = b_ih[col], biz = b_ih[col + OUTD], bin = b_ih[col + 2 * OUTD];
    const float bhr = b_hh[col], bhz = b_hh[col + OUTD], bhn = b_hh[col + 2 * OUTD];
    #pragma unroll
    for (int r = 0; r < 8; ++r) {
      int m = r + ((lane >> 4) << 3);
      size_t node = row0 + m;
      float hprev = hidden[node * OUTD + col];
      float rg = sigmoidf(ir[r] + bir + hr[r] + bhr);
      float zg = sigmoidf(iz[r] + biz + hz[r] + bhz);
      float ng = tanhf(in[r] + bin + rg * (hn[r] + bhn));
      float h  = (1.f - zg) * ng + zg * hprev;
      sHo[m * OUTD + col] = h;
      h_out[node * OUTD + col] = h;
    }
  }
  __syncthreads();

  // LayerNorm(h) -> packed-A layout in sGp (reuse), then q = LN(h) @ Wq + bq
  {
    const int row = lane & 15, base = (lane >> 4) * 128;
    float s = 0.f, s2 = 0.f;
    for (int c = 0; c < 128; ++c) { float v = sHo[row * OUTD + base + c]; s += v; s2 += v * v; }
    s  += __shfl_xor(s, 16, 32);
    s2 += __shfl_xor(s2, 16, 32);
    float mean = s * (1.f / OUTD);
    float var  = s2 * (1.f / OUTD) - mean * mean;
    float rstd = rsqrtf(var + 1e-5f);
    for (int c = 0; c < 128; ++c) {
      int col = base + c;
      float v = (sHo[row * OUTD + col] - mean) * rstd * g_out[col] + be_out[col];
      sGp[(col >> 5) * 512 + a_off(row, col & 31)] = (__bf16)v;
    }
  }
  __syncthreads();

  v8f aq = {};
  #pragma unroll
  for (int t = 0; t < 8; ++t)
    aq = wmma_bf16(load_Ap(sGp, t, lane), load_Bp(Wqp, 1, t, 0, lane), aq);
  const int col = lane & 15;
  const float bias = bq[col];
  #pragma unroll
  for (int r = 0; r < 8; ++r) {
    int m = r + ((lane >> 4) << 3);
    q_out[(row0 + m) * NA + col] = aq[r] + bias;
  }
}

// ---------------------------------------------------------------- launcher
extern "C" void kernel_launch(void* const* d_in, const int* in_sizes, int n_in,
                              void* d_out, int out_size, void* d_ws, size_t ws_size,
                              hipStream_t stream) {
  const float* inputs = (const float*)d_in[0];
  const float* hidden = (const float*)d_in[1];
  const float* eattr  = (const float*)d_in[2];
  const float* W0 = (const float*)d_in[3];  const float* b0  = (const float*)d_in[4];
  const float* g0 = (const float*)d_in[5];  const float* be0 = (const float*)d_in[6];
  const float* W1 = (const float*)d_in[7];  const float* b1  = (const float*)d_in[8];
  const float* g1 = (const float*)d_in[9];  const float* be1 = (const float*)d_in[10];
  const float* Wl = (const float*)d_in[11]; const float* bl  = (const float*)d_in[12];
  const float* Wr = (const float*)d_in[13]; const float* br  = (const float*)d_in[14];
  const float* We = (const float*)d_in[15]; const float* att = (const float*)d_in[16];
  const float* Wres = (const float*)d_in[17]; const float* b_gat = (const float*)d_in[18];
  const float* W_ih = (const float*)d_in[19]; const float* b_ih = (const float*)d_in[20];
  const float* W_hh = (const float*)d_in[21]; const float* b_hh = (const float*)d_in[22];
  const float* g_out = (const float*)d_in[23]; const float* be_out = (const float*)d_in[24];
  const float* Wq = (const float*)d_in[25]; const float* bq = (const float*)d_in[26];
  const int*   ei = (const int*)d_in[27];

  float* q_out = (float*)d_out;                 // N*16 (return order: q, h)
  float* h_out = q_out + (size_t)NN * NA;       // N*256

  // workspace layout
  float* x_l    = (float*)d_ws;
  float* x_r    = x_l  + (size_t)NN * OUTD;
  float* xres   = x_r  + (size_t)NN * OUTD;
  float* agg    = xres + (size_t)NN * OUTD;
  float* deg    = agg  + (size_t)NN * OUTD;
  float* loop3  = deg  + NN;
  float* smax   = loop3 + 3 * NN;
  float* den    = smax + NN;
  float* scoreb = den  + NN;                    // EF floats
  __bf16* W0p   = (__bf16*)(scoreb + EF);
  __bf16* W1p   = W0p   + DIN * HH;
  __bf16* Wlp   = W1p   + HH * HH;
  __bf16* Wrp   = Wlp   + HH * OUTD;
  __bf16* Wresp = Wrp   + HH * OUTD;
  __bf16* Wihp  = Wresp + HH * OUTD;
  __bf16* Whhp  = Wihp  + 3 * OUTD * OUTD;
  __bf16* Wqp   = Whhp  + 3 * OUTD * OUTD;
  __bf16* gatb  = Wqp   + OUTD * NA;

  // pack weights into fragment-major bf16 (one-time; L2-resident afterwards)
  auto pack = [&](const float* s, __bf16* t, int K, int Nc, int tr) {
    k_pack_B<<<(K * Nc + 255) / 256, 256, 0, stream>>>(s, t, K, Nc, tr);
  };
  pack(W0, W0p, DIN, HH, 0);    pack(W1, W1p, HH, HH, 0);
  pack(Wl, Wlp, HH, OUTD, 0);   pack(Wr, Wrp, HH, OUTD, 0);
  pack(Wres, Wresp, HH, OUTD, 0);
  pack(W_ih, Wihp, OUTD, 3 * OUTD, 1);   // used as x @ W^T
  pack(W_hh, Whhp, OUTD, 3 * OUTD, 1);
  pack(Wq, Wqp, OUTD, NA, 0);

  k_init<<<NN * OUTD / 256, 256, 0, stream>>>(agg, deg, loop3, smax, den);

  k_node<<<NN / 16, 32, 0, stream>>>(inputs, W0p, b0, g0, be0, W1p, b1, g1, be1,
                                     Wlp, bl, Wrp, br, Wresp, x_l, x_r, xres);

  k_edge_prep<<<(EE + 255) / 256, 256, 0, stream>>>(ei, eattr, deg, loop3);
  k_loop_fin<<<(NN + 255) / 256, 256, 0, stream>>>(loop3, deg);
  k_score<<<(EF + 7) / 8, 256, 0, stream>>>(ei, eattr, loop3, x_l, x_r, We, att, scoreb, smax);
  k_exp<<<(EF + 255) / 256, 256, 0, stream>>>(ei, scoreb, smax, den);
  k_agg<<<(EF + 7) / 8, 256, 0, stream>>>(ei, scoreb, den, x_l, agg);
  k_gat<<<NN, OUTD, 0, stream>>>(agg, xres, b_gat, gatb);

  k_gru<<<NN / 16, 32, 0, stream>>>(gatb, hidden, Wihp, b_ih, Whhp, b_hh,
                                    g_out, be_out, Wqp, bq, q_out, h_out);
}
